// MultiHeadAttention_28260884808257
// MI455X (gfx1250) — compile-verified, matrix operand order checked by reading, not run on hardware
//
#include <hip/hip_runtime.h>

// ---------------------------------------------------------------------------
// Problem constants (match reference)
// ---------------------------------------------------------------------------
#define BB   4
#define SS   2048
#define DD   512
#define HH   8
#define DH   64
#define MM   (BB * SS)      // 8192 rows for all projections
#define TOPK 32

typedef unsigned short u16;
typedef __attribute__((ext_vector_type(16))) __bf16 v16bf;
typedef __attribute__((ext_vector_type(8)))  float  v8f;

// ---------------------------------------------------------------------------
// bf16 helpers (round-to-nearest-even)
// ---------------------------------------------------------------------------
__device__ __forceinline__ u16 f2bf(float f) {
    union { float f; unsigned u; } v; v.f = f;
    unsigned u = v.u;
    u += 0x7fffu + ((u >> 16) & 1u);
    return (u16)(u >> 16);
}
__device__ __forceinline__ float bf2f(u16 h) {
    union { unsigned u; float f; } v; v.u = ((unsigned)h) << 16;
    return v.f;
}
__device__ __forceinline__ unsigned long long pack4bf(float4 f) {
    return (unsigned long long)f2bf(f.x)
         | ((unsigned long long)f2bf(f.y) << 16)
         | ((unsigned long long)f2bf(f.z) << 32)
         | ((unsigned long long)f2bf(f.w) << 48);
}

// ---------------------------------------------------------------------------
// GEMM:  Y[M,512] = X[M,512] * W[512,512] + bias
//   a_bf16 == 0 : X is fp32 (converted to bf16 while staging in LDS)
//   a_bf16 == 1 : X is bf16 (concat buffer)
//   mode   == 0 : write bf16, head-major layout [B,H,S,DH]
//   mode   == 1 : write fp32, row-major [M,512]  (final output)
// Tile: 128(M) x 64(N) x 32(K), 256 threads = 8 waves, each wave 32x32.
// ---------------------------------------------------------------------------
__global__ void __launch_bounds__(256)
gemm_bias(const float* __restrict__ Af, const u16* __restrict__ Abf,
          const float* __restrict__ W,  const float* __restrict__ bias,
          u16* __restrict__ Obf, float* __restrict__ Of,
          int a_bf16, int mode)
{
    __shared__ u16 sA[128 * 40];   // bf16 A tile, row stride 40 (16B-aligned rows)
    __shared__ u16 sB[64 * 40];    // bf16 B tile stored transposed: [n][k]

    const int tid  = threadIdx.x;
    const int wv   = tid >> 5;
    const int lane = tid & 31;
    const int half = lane >> 4;
    const int l16  = lane & 15;
    const int wm   = wv & 3;       // 4 row groups of 32
    const int wn   = wv >> 2;      // 2 col groups of 32
    const int n0   = blockIdx.x * 64;
    const int m0   = blockIdx.y * 128;

    v8f acc[2][2];
    #pragma unroll
    for (int i = 0; i < 2; ++i)
        #pragma unroll
        for (int j = 0; j < 2; ++j)
            acc[i][j] = v8f{};

    for (int kt = 0; kt < 16; ++kt) {
        const int k0 = kt * 32;

        // ---- stage A tile (128x32) into LDS as bf16 ----
        if (!a_bf16) {
            #pragma unroll
            for (int it = 0; it < 4; ++it) {
                int vid = tid + it * 256;            // 1024 float4 slots
                int row = vid >> 3;
                int c4  = (vid & 7) << 2;
                float4 f = *(const float4*)(Af + (size_t)(m0 + row) * DD + k0 + c4);
                *(unsigned long long*)&sA[row * 40 + c4] = pack4bf(f);
            }
        } else {
            #pragma unroll
            for (int it = 0; it < 2; ++it) {
                int vid = tid + it * 256;            // 512 x 8-elem slots
                int row = vid >> 2;
                int c8  = (vid & 3) << 3;
                __builtin_memcpy(&sA[row * 40 + c8],
                                 Abf + (size_t)(m0 + row) * DD + k0 + c8, 16);
            }
        }
        // ---- stage B tile (32x64) transposed into LDS as bf16 ----
        #pragma unroll
        for (int it = 0; it < 2; ++it) {
            int vid = tid + it * 256;                // 512 float4 slots
            int kr  = vid >> 4;
            int c4  = (vid & 15) << 2;
            float4 f = *(const float4*)(W + (size_t)(k0 + kr) * DD + n0 + c4);
            sB[(c4 + 0) * 40 + kr] = f2bf(f.x);
            sB[(c4 + 1) * 40 + kr] = f2bf(f.y);
            sB[(c4 + 2) * 40 + kr] = f2bf(f.z);
            sB[(c4 + 3) * 40 + kr] = f2bf(f.w);
        }
        __syncthreads();

        // ---- fragments per ISA 16-bit A 16x32 layout ----
        v16bf afr[2], bfr[2];
        #pragma unroll
        for (int i = 0; i < 2; ++i) {
            const u16* p = &sA[(wm * 32 + i * 16 + l16) * 40];
            __builtin_memcpy(&afr[i],               p + half * 8,      16);
            __builtin_memcpy((char*)&afr[i] + 16,   p + 16 + half * 8, 16);
        }
        #pragma unroll
        for (int j = 0; j < 2; ++j) {
            const u16* p = &sB[(wn * 32 + j * 16 + l16) * 40];
            __builtin_memcpy(&bfr[j],               p + half * 8,      16);
            __builtin_memcpy((char*)&bfr[j] + 16,   p + 16 + half * 8, 16);
        }
        #pragma unroll
        for (int i = 0; i < 2; ++i)
            #pragma unroll
            for (int j = 0; j < 2; ++j)
                acc[i][j] = __builtin_amdgcn_wmma_f32_16x16x32_bf16(
                    false, afr[i], false, bfr[j], (short)0, acc[i][j], false, false);
        __syncthreads();
    }

    // ---- epilogue ----
    #pragma unroll
    for (int i = 0; i < 2; ++i) {
        #pragma unroll
        for (int j = 0; j < 2; ++j) {
            const int gn = n0 + wn * 32 + j * 16 + l16;
            const float bb = bias[gn];
            #pragma unroll
            for (int r = 0; r < 8; ++r) {
                const int gm = m0 + wm * 32 + i * 16 + r + half * 8;
                const float val = acc[i][j][r] + bb;
                if (mode == 0) {
                    const int b = gm >> 11, s = gm & (SS - 1);
                    const int h = gn >> 6,  d = gn & (DH - 1);
                    Obf[((((size_t)(b * HH + h)) * SS + s) << 6) + d] = f2bf(val);
                } else {
                    Of[(size_t)gm * DD + gn] = val;
                }
            }
        }
    }
}

// ---------------------------------------------------------------------------
// Fused attention core: one workgroup = one (b,h) x 16-query strip.
//
// Phase 1: logits[16][2048] = (Q16 . K^T)/8 via WMMA, fp32 into 128 KB of
//          dynamic LDS (CDNA5 allows up to 320 KB per workgroup).  8 waves
//          each own 16 of the 128 key-blocks; fragments are loaded straight
//          from the bf16 head-layout Qh/Kh rows (dh=64 -> two K-chunks).
// Phase 2: one wave per query row (2 rows each): top-32 by iterated
//          wave-argmax (earlier index wins ties, matching lax.top_k),
//          L1-renormalized softmax over the survivors only (the dense
//          softmax denominator cancels), sparse V-gather, and a single
//          coalesced write of the dense sparse-w row.  Logits never
//          touch HBM.
// ---------------------------------------------------------------------------
__global__ void __launch_bounds__(256)
fused_attn(const u16* __restrict__ Qh, const u16* __restrict__ Kh,
           const u16* __restrict__ Vh, float* __restrict__ Wout,
           u16* __restrict__ concat)
{
    extern __shared__ float slog[];               // [16][2048] fp32 = 128 KB

    const int tid  = threadIdx.x;
    const int wvid = tid >> 5;
    const int lane = tid & 31;
    const int half = lane >> 4;
    const int l16  = lane & 15;
    const int bz   = blockIdx.y;                  // b*H + h
    const int q0   = blockIdx.x * 16;             // query strip base
    const int b    = bz >> 3;
    const int h    = bz & 7;

    // ---- phase 1: logits into LDS ----
    v16bf afr[2];
    {
        const u16* qp = Qh + ((size_t)bz * SS + (q0 + l16)) * DH;
        #pragma unroll
        for (int c = 0; c < 2; ++c) {
            __builtin_memcpy(&afr[c],             qp + c * 32 + half * 8,      16);
            __builtin_memcpy((char*)&afr[c] + 16, qp + c * 32 + 16 + half * 8, 16);
        }
    }
    for (int jb = wvid; jb < SS / 16; jb += 8) {
        const int j0 = jb * 16;
        const u16* kp = Kh + ((size_t)bz * SS + (j0 + l16)) * DH;
        // prefetch next key block for this wave (global_prefetch_b8 path)
        if (jb + 8 < SS / 16)
            __builtin_prefetch(Kh + ((size_t)bz * SS + (j0 + 128 + l16)) * DH, 0, 1);

        v16bf bfr[2];
        #pragma unroll
        for (int c = 0; c < 2; ++c) {
            __builtin_memcpy(&bfr[c],             kp + c * 32 + half * 8,      16);
            __builtin_memcpy((char*)&bfr[c] + 16, kp + c * 32 + 16 + half * 8, 16);
        }
        v8f acc = v8f{};
        #pragma unroll
        for (int c = 0; c < 2; ++c)
            acc = __builtin_amdgcn_wmma_f32_16x16x32_bf16(
                false, afr[c], false, bfr[c], (short)0, acc, false, false);
        #pragma unroll
        for (int r = 0; r < 8; ++r)
            slog[(r + half * 8) * SS + j0 + l16] = acc[r] * 0.125f;
    }
    __syncthreads();

    // ---- phase 2: top-k + renorm + sparse attention (2 rows per wave) ----
    const u16* Vbase = Vh + (size_t)bz * SS * DH;
    #pragma unroll 1
    for (int rr = 0; rr < 2; ++rr) {
        const int qloc = wvid * 2 + rr;
        const int q    = q0 + qloc;
        const float* lrow = slog + (size_t)qloc * SS;

        float vals[64];
        #pragma unroll
        for (int t = 0; t < 64; ++t)
            vals[t] = lrow[t * 32 + lane];

        // row max (for exp stability)
        float m = vals[0];
        #pragma unroll
        for (int t = 1; t < 64; ++t) m = fmaxf(m, vals[t]);
        #pragma unroll
        for (int off = 16; off > 0; off >>= 1)
            m = fmaxf(m, __shfl_xor(m, off));

        unsigned long long mask = ~0ull;          // unselected bits per lane
        float sumexp = 0.0f, acc0 = 0.0f, acc1 = 0.0f;

        for (int it = 0; it < TOPK; ++it) {
            float bv = -__builtin_inff();
            int bidx = 0x7fffffff;
            #pragma unroll
            for (int t = 0; t < 64; ++t) {
                if (mask & (1ull << t)) {
                    const float v = vals[t];
                    const int  id = t * 32 + lane;
                    if (v > bv || (v == bv && id < bidx)) { bv = v; bidx = id; }
                }
            }
            // wave argmax (butterfly -> all lanes agree)
            #pragma unroll
            for (int off = 16; off > 0; off >>= 1) {
                const float ov = __shfl_xor(bv, off);
                const int   oi = __shfl_xor(bidx, off);
                if (ov > bv || (ov == bv && oi < bidx)) { bv = ov; bidx = oi; }
            }
            if ((bidx & 31) == lane) mask &= ~(1ull << (bidx >> 5));

            const float e = __expf(bv - m);       // uniform across wave
            sumexp += e;
            const u16* vr = Vbase + (size_t)bidx * DH;
            acc0 += e * bf2f(vr[lane]);
            acc1 += e * bf2f(vr[lane + 32]);
        }

        const float inv = 1.0f / sumexp;

        // single coalesced write of the dense sparse-w row
        float* wrow = Wout + ((size_t)bz * SS + q) * SS;
        #pragma unroll
        for (int t = 0; t < 64; ++t) {
            const float w = ((mask >> t) & 1ull) ? 0.0f
                                                 : __expf(vals[t] - m) * inv;
            wrow[t * 32 + lane] = w;
        }

        // attention output -> bf16 concat [B,S,D]
        u16* cp = concat + ((size_t)(b * SS + q)) * DD + h * DH;
        cp[lane]      = f2bf(acc0 * inv);
        cp[lane + 32] = f2bf(acc1 * inv);
    }
}

// ---------------------------------------------------------------------------
// Launch
// inputs: 0=v 1=k 2=q 3=Wq 4=bq 5=Wk 6=bk 7=Wv 8=bv 9=Wo 10=bo 11=top_k
// d_out:  [out fp32 (B*S*D)] ++ [w fp32 (B*H*S*S)]
// d_ws:   Qh | Kh | Vh | concat  (bf16, each B*H*S*DH = 4194304 elems = 8 MB)
// ---------------------------------------------------------------------------
extern "C" void kernel_launch(void* const* d_in, const int* in_sizes, int n_in,
                              void* d_out, int out_size, void* d_ws, size_t ws_size,
                              hipStream_t stream)
{
    const float* v  = (const float*)d_in[0];
    const float* k  = (const float*)d_in[1];
    const float* q  = (const float*)d_in[2];
    const float* Wq = (const float*)d_in[3];
    const float* bq = (const float*)d_in[4];
    const float* Wk = (const float*)d_in[5];
    const float* bk = (const float*)d_in[6];
    const float* Wv = (const float*)d_in[7];
    const float* bv = (const float*)d_in[8];
    const float* Wo = (const float*)d_in[9];
    const float* bo = (const float*)d_in[10];

    float* out0 = (float*)d_out;                       // [B,S,D]
    float* wout = out0 + (size_t)BB * SS * DD;         // [B,H,S,S]

    const size_t HE = (size_t)BB * HH * SS * DH;       // 4,194,304 elems
    u16* Qh     = (u16*)d_ws;
    u16* Kh     = Qh + HE;
    u16* Vh     = Kh + HE;
    u16* concat = Vh + HE;

    const dim3 gemm_grid(DD / 64, MM / 128);           // (8, 64)
    const int  fused_lds = 16 * SS * (int)sizeof(float); // 128 KB dynamic LDS

    // allow >64 KB dynamic LDS (host-side config; graph-capture safe,
    // idempotent, called unconditionally every launch)
    (void)hipFuncSetAttribute(reinterpret_cast<const void*>(fused_attn),
                              hipFuncAttributeMaxDynamicSharedMemorySize,
                              fused_lds);

    // Q/K/V projections (fp32 in -> bf16 head layout)
    gemm_bias<<<gemm_grid, 256, 0, stream>>>(q, nullptr, Wq, bq, Qh, nullptr, 0, 0);
    gemm_bias<<<gemm_grid, 256, 0, stream>>>(k, nullptr, Wk, bk, Kh, nullptr, 0, 0);
    gemm_bias<<<gemm_grid, 256, 0, stream>>>(v, nullptr, Wv, bv, Vh, nullptr, 0, 0);

    // fused logits + top-32 + renorm + sparse attention (w written once)
    fused_attn<<<dim3(SS / 16, BB * HH), 256, fused_lds, stream>>>(
        Qh, Kh, Vh, wout, concat);

    // output projection (bf16 concat in -> fp32 out)
    gemm_bias<<<gemm_grid, 256, 0, stream>>>(nullptr, concat, Wo, bo, nullptr, out0, 1, 1);

    (void)in_sizes; (void)n_in; (void)out_size; (void)ws_size;
}
